// SelfAttention_944892805585
// MI455X (gfx1250) — matrix operand order repaired and checked by reading, not compiled, not run on hardware
//
#include <hip/hip_runtime.h>
#include <hip/hip_bf16.h>

#define BATCH 4
#define SEQ   4096
#define DM    256
#define BL    (BATCH*SEQ)
#define TEMP_INV 0.0625f   // 1/sqrt(256)
#define LN_EPS   1e-6f

typedef __attribute__((ext_vector_type(16))) __bf16 v16bf;
typedef __attribute__((ext_vector_type(8)))  __bf16 v8bf;
typedef __attribute__((ext_vector_type(8)))  float  v8f;
typedef __attribute__((ext_vector_type(4)))  float  v4f;

// ---------------------------------------------------------------------------
// WMMA fragment loaders (contiguous k order, same in A and B, wave32).
// A 16x32 bf16: lane m = lane&15; halves e=0..15 -> K = hi*16+e, hi = lane>>4.
// B 32x16 bf16: lane n = lane&15; halves e=0..15 -> K = hi*16+e.
// C/D 16x16 f32: row = (lane>>4)*8 + r, col = lane&15.
// ---------------------------------------------------------------------------

__device__ __forceinline__ v16bf load_a_bf16(const __bf16* A, long ld, long row0, long k0) {
  const int lane = threadIdx.x & 31;
  const __bf16* p = A + (row0 + (lane & 15)) * ld + k0 + (long)(lane >> 4) * 16;
  v8bf a0 = *(const v8bf*)(p);
  v8bf a1 = *(const v8bf*)(p + 8);
  v16bf f;
#pragma unroll
  for (int e = 0; e < 8; ++e) { f[e] = a0[e]; f[8 + e] = a1[e]; }
  return f;
}

__device__ __forceinline__ v16bf load_a_f32(const float* A, long ld, long row0, long k0) {
  const int lane = threadIdx.x & 31;
  const float* p = A + (row0 + (lane & 15)) * ld + k0 + (long)(lane >> 4) * 16;
  v4f a0 = *(const v4f*)(p);
  v4f a1 = *(const v4f*)(p + 4);
  v4f a2 = *(const v4f*)(p + 8);
  v4f a3 = *(const v4f*)(p + 12);
  v16bf f;
#pragma unroll
  for (int e = 0; e < 4; ++e) {
    f[e]      = (__bf16)a0[e];
    f[4 + e]  = (__bf16)a1[e];
    f[8 + e]  = (__bf16)a2[e];
    f[12 + e] = (__bf16)a3[e];
  }
  return f;
}

__device__ __forceinline__ v16bf load_b_bf16(const __bf16* B, long ld, long n0, long k0) {
  const int lane = threadIdx.x & 31;
  const __bf16* p = B + (n0 + (lane & 15)) * ld + k0 + (long)(lane >> 4) * 16;
  v8bf b0 = *(const v8bf*)(p);
  v8bf b1 = *(const v8bf*)(p + 8);
  v16bf f;
#pragma unroll
  for (int e = 0; e < 8; ++e) { f[e] = b0[e]; f[8 + e] = b1[e]; }
  return f;
}

__device__ __forceinline__ v16bf load_b_f32(const float* B, long ld, long n0, long k0) {
  const int lane = threadIdx.x & 31;
  const float* p = B + (n0 + (lane & 15)) * ld + k0 + (long)(lane >> 4) * 16;
  v4f b0 = *(const v4f*)(p);
  v4f b1 = *(const v4f*)(p + 4);
  v4f b2 = *(const v4f*)(p + 8);
  v4f b3 = *(const v4f*)(p + 12);
  v16bf f;
#pragma unroll
  for (int e = 0; e < 4; ++e) {
    f[e]      = (__bf16)b0[e];
    f[4 + e]  = (__bf16)b1[e];
    f[8 + e]  = (__bf16)b2[e];
    f[12 + e] = (__bf16)b3[e];
  }
  return f;
}

__device__ __forceinline__ v8f wmma_bf16(v16bf a, v16bf b, v8f c) {
  return __builtin_amdgcn_wmma_f32_16x16x32_bf16(
      /*neg_a=*/false, a, /*neg_b=*/false, b,
      /*c_mod=*/(short)0, c, /*reuse_a=*/false, /*reuse_b=*/false);
}

// ---------------------------------------------------------------------------
// K1: projections. qh = (q @ Wq^T)/sqrt(d)  (bf16), kh = k @ Wk^T (bf16),
//     vht = (v @ Wv^T)^T per batch (bf16, [DM x SEQ]).
// One wave -> 16x64 strip. blockIdx.y selects op.
// ---------------------------------------------------------------------------
__global__ void proj_kernel(const float* __restrict__ q, const float* __restrict__ k,
                            const float* __restrict__ v,
                            const float* __restrict__ wq, const float* __restrict__ wk,
                            const float* __restrict__ wv,
                            __bf16* __restrict__ qh, __bf16* __restrict__ kh,
                            __bf16* __restrict__ vht) {
  const int op  = blockIdx.y;
  const int wid = blockIdx.x * (blockDim.x >> 5) + (threadIdx.x >> 5);
  const long row0 = (long)(wid >> 2) * 16;   // 16-row tile in [BL, DM]
  const long n0   = (long)(wid & 3) * 64;    // 64-col strip of DM
  const float* src = (op == 0) ? q : (op == 1) ? k : v;
  const float* w   = (op == 0) ? wq : (op == 1) ? wk : wv;

  v8f acc[4] = {};
  for (int kk = 0; kk < DM; kk += 32) {
    v16bf a = load_a_f32(src, DM, row0, kk);
#pragma unroll
    for (int j = 0; j < 4; ++j) {
      v16bf b = load_b_f32(w, DM, n0 + j * 16, kk);
      acc[j] = wmma_bf16(a, b, acc[j]);
    }
  }
  const int lane = threadIdx.x & 31;
  const int n = lane & 15, mb = (lane >> 4) * 8;
  if (op == 2) {
    // transposed store: vht[b][col][pos]
#pragma unroll
    for (int j = 0; j < 4; ++j) {
      const long col = n0 + j * 16 + n;
#pragma unroll
      for (int r = 0; r < 8; ++r) {
        const long grow = row0 + mb + r;
        const long b    = grow >> 12;      // /SEQ
        const long pos  = grow & (SEQ - 1);
        vht[(b * DM + col) * (long)SEQ + pos] = (__bf16)acc[j][r];
      }
    }
  } else {
    __bf16* dst = (op == 0) ? qh : kh;
    const float scale = (op == 0) ? TEMP_INV : 1.0f;
#pragma unroll
    for (int j = 0; j < 4; ++j) {
      const long col = n0 + j * 16 + n;
#pragma unroll
      for (int r = 0; r < 8; ++r)
        dst[(row0 + mb + r) * DM + col] = (__bf16)(acc[j][r] * scale);
    }
  }
}

// ---------------------------------------------------------------------------
// K2: scores = qh @ kh^T (per batch), f32 logits -> attn region of d_out.
// One wave -> 16 queries x 128 keys (8 accumulators) so each A fragment
// feeds 8 WMMAs; kh strips stay L2-resident (2 MB/batch vs 192 MB L2).
// ---------------------------------------------------------------------------
__global__ void scores_kernel(const __bf16* __restrict__ qh, const __bf16* __restrict__ kh,
                              float* __restrict__ attn) {
  const int wid = blockIdx.x * (blockDim.x >> 5) + (threadIdx.x >> 5);
  const int ks  = wid & 31;          // key strip of 128 (SEQ/128)
  const int rem = wid >> 5;
  const int qt  = rem & 255;         // query tile of 16 (SEQ/16)
  const int b   = rem >> 8;
  const long row0 = (long)qt * 16;
  const long n0   = (long)ks * 128;
  const __bf16* A  = qh + (long)b * SEQ * DM;
  const __bf16* Bm = kh + (long)b * SEQ * DM;

  v8f acc[8] = {};
  for (int kk = 0; kk < DM; kk += 32) {
    v16bf a = load_a_bf16(A, DM, row0, kk);
#pragma unroll
    for (int j = 0; j < 8; ++j) {
      v16bf bb = load_b_bf16(Bm, DM, n0 + j * 16, kk);
      acc[j] = wmma_bf16(a, bb, acc[j]);
    }
  }
  const int lane = threadIdx.x & 31;
  const int n = lane & 15, mb = (lane >> 4) * 8;
  float* dst = attn + (long)b * SEQ * SEQ;
#pragma unroll
  for (int j = 0; j < 8; ++j)
#pragma unroll
    for (int r = 0; r < 8; ++r)
      dst[(row0 + mb + r) * (long)SEQ + n0 + j * 16 + n] = acc[j][r];
}

// ---------------------------------------------------------------------------
// K3: row softmax in place over attn (row length SEQ). One block per row.
// ---------------------------------------------------------------------------
__global__ void softmax_kernel(float* __restrict__ attn) {
  float* p = attn + (long)blockIdx.x * SEQ;
  const int t = threadIdx.x;           // 256 threads
  __shared__ float red[256];

  float x[16];
  float m = -3.402823466e38f;
#pragma unroll
  for (int i = 0; i < 16; ++i) { x[i] = p[t + i * 256]; m = fmaxf(m, x[i]); }
  red[t] = m; __syncthreads();
  for (int s = 128; s > 0; s >>= 1) {
    if (t < s) red[t] = fmaxf(red[t], red[t + s]);
    __syncthreads();
  }
  m = red[0]; __syncthreads();

  float s = 0.f;
#pragma unroll
  for (int i = 0; i < 16; ++i) { x[i] = __expf(x[i] - m); s += x[i]; }
  red[t] = s; __syncthreads();
  for (int st = 128; st > 0; st >>= 1) {
    if (t < st) red[t] += red[t + st];
    __syncthreads();
  }
  const float inv = 1.0f / red[0];
#pragma unroll
  for (int i = 0; i < 16; ++i) p[t + i * 256] = x[i] * inv;
}

// ---------------------------------------------------------------------------
// K4: o = attn @ vh. Dominant-traffic kernel: each wave now produces the FULL
// 16x256 output tile (16 v8f accumulators = 128 VGPRs) so every attn element
// (268 MB, f32->bf16 on load) is read exactly ONCE from memory. vht B-operand
// (2 MB/batch) is L2-resident.
// ---------------------------------------------------------------------------
__global__ void av_kernel(const float* __restrict__ attn, const __bf16* __restrict__ vht,
                          __bf16* __restrict__ o) {
  const int wid = blockIdx.x * (blockDim.x >> 5) + (threadIdx.x >> 5);
  const int qt  = wid & 255;           // SEQ/16 query tiles
  const int b   = wid >> 8;
  const long row0 = (long)qt * 16;
  const float*  A  = attn + (long)b * SEQ * SEQ;
  const __bf16* Bm = vht + (long)b * DM * (long)SEQ;

  v8f acc[16] = {};
  for (int kk = 0; kk < SEQ; kk += 32) {
    v16bf a = load_a_f32(A, SEQ, row0, kk);
#pragma unroll
    for (int j = 0; j < 16; ++j) {
      v16bf bb = load_b_bf16(Bm, SEQ, (long)j * 16, kk);
      acc[j] = wmma_bf16(a, bb, acc[j]);
    }
  }
  const int lane = threadIdx.x & 31;
  const int n = lane & 15, mb = (lane >> 4) * 8;
#pragma unroll
  for (int j = 0; j < 16; ++j)
#pragma unroll
    for (int r = 0; r < 8; ++r)
      o[((long)b * SEQ + row0 + mb + r) * DM + j * 16 + n] = (__bf16)acc[j][r];
}

// ---------------------------------------------------------------------------
// K5: out = LN(o @ Wfc^T + residual). 128 threads (4 waves) per 16-row tile;
//     GEMM strips land in a 16x256 LDS tile, then fused LayerNorm.
// ---------------------------------------------------------------------------
__global__ void fc_ln_kernel(const __bf16* __restrict__ o, const float* __restrict__ wfc,
                             const float* __restrict__ resid, const float* __restrict__ gamma,
                             const float* __restrict__ beta, float* __restrict__ out) {
  __shared__ float y[16][DM];
  __shared__ float mu_s[16], rs_s[16];
  const long row0 = (long)blockIdx.x * 16;
  const long n0   = (long)(threadIdx.x >> 5) * 64;

  v8f acc[4] = {};
  for (int kk = 0; kk < DM; kk += 32) {
    v16bf a = load_a_bf16(o, DM, row0, kk);
#pragma unroll
    for (int j = 0; j < 4; ++j) {
      v16bf bb = load_b_f32(wfc, DM, n0 + j * 16, kk);
      acc[j] = wmma_bf16(a, bb, acc[j]);
    }
  }
  const int lane = threadIdx.x & 31;
  const int n = lane & 15, mb = (lane >> 4) * 8;
#pragma unroll
  for (int j = 0; j < 4; ++j)
#pragma unroll
    for (int r = 0; r < 8; ++r) {
      const int rr = mb + r;
      const int cc = (int)n0 + j * 16 + n;
      y[rr][cc] = acc[j][r] + resid[(row0 + rr) * DM + cc];
    }
  __syncthreads();

  const int t = threadIdx.x;
  if (t < 16) {
    float s = 0.f;
    for (int c = 0; c < DM; ++c) s += y[t][c];
    const float mu = s * (1.0f / DM);
    float v2 = 0.f;
    for (int c = 0; c < DM; ++c) { const float d = y[t][c] - mu; v2 += d * d; }
    mu_s[t] = mu;
    rs_s[t] = rsqrtf(v2 * (1.0f / DM) + LN_EPS);
  }
  __syncthreads();

  for (int idx = t; idx < 16 * DM; idx += blockDim.x) {
    const int rr = idx >> 8;
    const int cc = idx & (DM - 1);
    out[(row0 + rr) * DM + cc] = (y[rr][cc] - mu_s[rr]) * rs_s[rr] * gamma[cc] + beta[cc];
  }
}

// ---------------------------------------------------------------------------
extern "C" void kernel_launch(void* const* d_in, const int* in_sizes, int n_in,
                              void* d_out, int out_size, void* d_ws, size_t ws_size,
                              hipStream_t stream) {
  const float* q     = (const float*)d_in[0];
  const float* k     = (const float*)d_in[1];
  const float* v     = (const float*)d_in[2];
  const float* wq    = (const float*)d_in[3];
  const float* wk    = (const float*)d_in[4];
  const float* wv    = (const float*)d_in[5];
  const float* wfc   = (const float*)d_in[6];
  const float* gamma = (const float*)d_in[7];
  const float* beta  = (const float*)d_in[8];

  float* out  = (float*)d_out;
  float* attn = out + (long)BL * DM;       // attn region of d_out: [B,1,L,L]

  char* ws = (char*)d_ws;
  const size_t SZ = (size_t)BL * DM * sizeof(__bf16);   // 8 MiB per tensor
  __bf16* qh  = (__bf16*)(ws);
  __bf16* kh  = (__bf16*)(ws + SZ);
  __bf16* vht = (__bf16*)(ws + 2 * SZ);
  __bf16* o   = (__bf16*)(ws + 3 * SZ);

  // K1: 3 ops x (BL/16 row tiles x 4 strips) = 3 x 4096 waves; 8 waves/block
  dim3 g1(512, 3);
  proj_kernel<<<g1, 256, 0, stream>>>(q, k, v, wq, wk, wv, qh, kh, vht);
  // K2: 4 * 256 * 32 = 32768 waves (16x128 tiles)
  scores_kernel<<<4096, 256, 0, stream>>>(qh, kh, attn);
  // K3: one block per row
  softmax_kernel<<<BL, 256, 0, stream>>>(attn);
  // K4: 4 * 256 = 1024 waves (16x256 tiles, attn read exactly once)
  av_kernel<<<128, 256, 0, stream>>>(attn, vht, o);
  // K5: BL/16 tiles, 128 threads each
  fc_ln_kernel<<<1024, 128, 0, stream>>>(o, wfc, q, gamma, beta, out);
}
